// TransformerGCN_71167608094895
// MI455X (gfx1250) — matrix operand order repaired and checked by reading, not compiled, not run on hardware
//
#include <hip/hip_runtime.h>
#include <hip/hip_bf16.h>
#include <stdint.h>

#define HDIM   512   // H*D
#define NHEAD  4
#define DHEAD  128
#define NGRAPH 64
#define NOUT   12
#define KC     128   // K-chunk staged in LDS per double-buffer slot

typedef __attribute__((ext_vector_type(16))) __bf16       bf16x16;
typedef __attribute__((ext_vector_type(8)))  float        f32x8;
typedef __attribute__((ext_vector_type(4)))  unsigned int u32x4;
typedef __attribute__((ext_vector_type(4)))  float        f32x4;

union FragBF {
    bf16x16 v;
    u32x4   u[2];
};

__device__ __forceinline__ unsigned short f2bf(float f) {
    __bf16 b = (__bf16)f;
    unsigned short u;
    __builtin_memcpy(&u, &b, 2);
    return u;
}

// CDNA5 async global->LDS copy (ASYNCcnt-tracked) + its wait.
__device__ __forceinline__ void async_copy_b128(const void* gptr, void* lptr) {
    unsigned lds = (unsigned)(unsigned long long)lptr;   // low 32 bits = LDS offset
    asm volatile("global_load_async_to_lds_b128 %0, %1, off"
                 :: "v"(lds), "v"(gptr) : "memory");
}
__device__ __forceinline__ void wait_async0() {
    asm volatile("s_wait_asynccnt 0x0" ::: "memory");
}

// ---------------------------------------------------------------------------
// GEMM: C[N x 512] = A[N x K](bf16) * W[K x 512] + bias, W given as WT[512 x K].
// Block = 256 threads = 8 waves -> 128x64 output tile; wave -> 16x64 strip via
// 4 accumulators of v_wmma_f32_16x16x32_bf16. B panel (shared by all 8 waves)
// is staged in LDS with double-buffered async copies; A is register
// double-buffered. Requires N % 16 == 0 (N = 20000 -> ok).
// ---------------------------------------------------------------------------
__global__ void __launch_bounds__(256)
gemm_bf16_wmma(const unsigned short* __restrict__ A,   // N x K row-major (bf16 bits)
               const unsigned short* __restrict__ WT,  // 512 x K row-major (bf16 bits)
               const float* __restrict__ bias,         // 512
               float* __restrict__ C,                  // N x 512 f32
               int K, int nRowTiles)
{
    __shared__ unsigned short smB[2][64][KC + 8];      // 2 x 64 x 136 bf16 = 34,816 B

    const int tid  = threadIdx.x;
    const int lane = tid & 31;
    const int wave = tid >> 5;
    const int rowTile = blockIdx.x * 8 + wave;
    const bool active = rowTile < nRowTiles;           // inactive waves still stage+barrier
    const int row0 = active ? rowTile * 16 : 0;
    const int col0 = blockIdx.y * 64;
    const int m    = lane & 15;
    const int hi   = lane >> 4;                        // 0 / 1
    const int kA   = hi * 8;                           // A: upper lanes hold K+8..15 / K+24..31
    const int kB   = hi * 8;                           // B: upper lanes hold K+8..15 / K+24..31

    // cooperative staging: thread t -> B row (t>>2), 16B parts {t&3, +4, +8, +12}
    const int srow  = tid >> 2;
    const int spart = tid & 3;
    const unsigned short* gRow = WT + (size_t)(col0 + srow) * K;

    auto stage = [&](int kc, int buf) {
        #pragma unroll
        for (int w = 0; w < 4; ++w) {
            int word = spart + w * 4;                  // 16B units within 256B row chunk
            async_copy_b128(gRow + kc + word * 8, &smB[buf][srow][word * 8]);
        }
    };

    const unsigned short* aRow = A + (size_t)(row0 + m) * K;

    f32x8 acc0 = {}, acc1 = {}, acc2 = {}, acc3 = {};

    stage(0, 0);
    wait_async0();
    __syncthreads();

    FragBF a;
    a.u[0] = *(const u32x4*)(aRow + kA);
    a.u[1] = *(const u32x4*)(aRow + 16 + kA);

    for (int kc = 0; kc < K; kc += KC) {
        const int cur = (kc / KC) & 1;
        if (kc + KC < K) {
            stage(kc + KC, cur ^ 1);                   // overlap DMA with WMMA below
            __builtin_prefetch(aRow + kc + KC, 0, 3);  // WGP-scope prefetch of next A
        }

        #pragma unroll
        for (int k0 = 0; k0 < KC; k0 += 32) {
            // register double-buffer: issue next A fragment loads before the WMMAs
            int kn = kc + k0 + 32;
            if (kn >= K) kn = 0;                       // redundant but in-bounds
            FragBF an;
            an.u[0] = *(const u32x4*)(aRow + kn + kA);
            an.u[1] = *(const u32x4*)(aRow + kn + 16 + kA);

            const unsigned short* bp0 = &smB[cur][ 0 + m][k0 + kB];
            const unsigned short* bp1 = &smB[cur][16 + m][k0 + kB];
            const unsigned short* bp2 = &smB[cur][32 + m][k0 + kB];
            const unsigned short* bp3 = &smB[cur][48 + m][k0 + kB];
            FragBF b0, b1, b2, b3;
            b0.u[0] = *(const u32x4*)(bp0);  b0.u[1] = *(const u32x4*)(bp0 + 16);
            b1.u[0] = *(const u32x4*)(bp1);  b1.u[1] = *(const u32x4*)(bp1 + 16);
            b2.u[0] = *(const u32x4*)(bp2);  b2.u[1] = *(const u32x4*)(bp2 + 16);
            b3.u[0] = *(const u32x4*)(bp3);  b3.u[1] = *(const u32x4*)(bp3 + 16);

            acc0 = __builtin_amdgcn_wmma_f32_16x16x32_bf16(false, a.v, false, b0.v, (short)0, acc0, false, false);
            acc1 = __builtin_amdgcn_wmma_f32_16x16x32_bf16(false, a.v, false, b1.v, (short)0, acc1, false, false);
            acc2 = __builtin_amdgcn_wmma_f32_16x16x32_bf16(false, a.v, false, b2.v, (short)0, acc2, false, false);
            acc3 = __builtin_amdgcn_wmma_f32_16x16x32_bf16(false, a.v, false, b3.v, (short)0, acc3, false, false);
            a = an;
        }

        wait_async0();                                 // own async copies landed
        __syncthreads();                               // everyone's copies landed
    }

    if (!active) return;

    const int col = col0 + m;
    const int r0  = row0 + hi * 8;
    const float bb0 = bias[col], bb1 = bias[col + 16], bb2 = bias[col + 32], bb3 = bias[col + 48];
    #pragma unroll
    for (int r = 0; r < 8; ++r) {
        float* crow = C + (size_t)(r0 + r) * HDIM;
        crow[col]      = acc0[r] + bb0;
        crow[col + 16] = acc1[r] + bb1;
        crow[col + 32] = acc2[r] + bb2;
        crow[col + 48] = acc3[r] + bb3;
    }
}

// ---------------------------------------------------------------------------
// Helpers: conversion / transpose / fills
// ---------------------------------------------------------------------------
__global__ void __launch_bounds__(256)
cvt_bf16_kernel(const float* __restrict__ in, unsigned short* __restrict__ outb,
                float* __restrict__ outf, size_t n, int doRelu)
{
    size_t i = (size_t)blockIdx.x * 256 + threadIdx.x;
    if (i >= n) return;
    float v = in[i];
    if (doRelu) v = fmaxf(v, 0.0f);
    if (outb) outb[i] = f2bf(v);
    if (outf) outf[i] = v;
}

__global__ void __launch_bounds__(256)
transpose_bf16_kernel(const float* __restrict__ W, unsigned short* __restrict__ WT,
                      int K, int M)   // W: K x M f32  ->  WT: M x K bf16
{
    int i = blockIdx.x * 256 + threadIdx.x;
    if (i >= K * M) return;
    int m = i / K, k = i % K;
    WT[i] = f2bf(W[(size_t)k * M + m]);
}

__global__ void __launch_bounds__(256)
fill_f32_kernel(float* __restrict__ p, float v, size_t n)
{
    size_t i = (size_t)blockIdx.x * 256 + threadIdx.x;
    if (i < n) p[i] = v;
}

// ---------------------------------------------------------------------------
// Edge phase
// ---------------------------------------------------------------------------
__device__ __forceinline__ void atomicMaxF(float* addr, float val) {
    if (val >= 0.0f) atomicMax((int*)addr, __float_as_int(val));
    else             atomicMin((unsigned int*)addr, __float_as_uint(val));
}

// one wave per edge: logits[e][h] = q[dst,h,:] . k[src,h,:] / sqrt(D)
__global__ void __launch_bounds__(256)
edge_logits_kernel(const float* __restrict__ q, const float* __restrict__ k,
                   const int* __restrict__ src, const int* __restrict__ dst,
                   float* __restrict__ logit, int E_)
{
    int e = blockIdx.x * 8 + (threadIdx.x >> 5);
    if (e >= E_) return;
    int lane = threadIdx.x & 31;
    int s = src[e], d = dst[e];
    const f32x4* qp = (const f32x4*)(q + (size_t)d * HDIM);
    const f32x4* kp = (const f32x4*)(k + (size_t)s * HDIM);
    #pragma unroll
    for (int h = 0; h < NHEAD; ++h) {
        f32x4 qa = qp[h * 32 + lane];
        f32x4 ka = kp[h * 32 + lane];
        float p = qa.x * ka.x + qa.y * ka.y + qa.z * ka.z + qa.w * ka.w;
        #pragma unroll
        for (int off = 16; off > 0; off >>= 1) p += __shfl_xor(p, off, 32);
        if (lane == 0) logit[(size_t)e * NHEAD + h] = p * 0.08838834764831845f; // 1/sqrt(128)
    }
}

__global__ void __launch_bounds__(256)
seg_max_kernel(const float* __restrict__ logit, const int* __restrict__ dst,
               float* __restrict__ mbuf, int EH)
{
    int i = blockIdx.x * 256 + threadIdx.x;
    if (i >= EH) return;
    int e = i >> 2, h = i & 3;
    atomicMaxF(&mbuf[(size_t)dst[e] * NHEAD + h], logit[i]);
}

__global__ void __launch_bounds__(256)
exp_denom_kernel(float* __restrict__ logit, const int* __restrict__ dst,
                 const float* __restrict__ mbuf, float* __restrict__ denom, int EH)
{
    int i = blockIdx.x * 256 + threadIdx.x;
    if (i >= EH) return;
    int e = i >> 2, h = i & 3;
    int d = dst[e];
    float v = __expf(logit[i] - mbuf[(size_t)d * NHEAD + h]);
    logit[i] = v;                      // logit buffer now holds ex
    atomicAdd(&denom[(size_t)d * NHEAD + h], v);
}

// thread per (edge, 4 contiguous dims): agg[dst] += alpha * v[src]
__global__ void __launch_bounds__(256)
scatter_kernel(const float* __restrict__ ex, const float* __restrict__ denom,
               const float* __restrict__ vb, const int* __restrict__ src,
               const int* __restrict__ dst, float* __restrict__ agg, long long total)
{
    long long i = (long long)blockIdx.x * 256 + threadIdx.x;
    if (i >= total) return;
    int e = (int)(i >> 7);
    int c = (int)(i & 127);            // chunk of 4 dims
    int h = c >> 5;
    int d = dst[e], s = src[e];
    float alpha = ex[(size_t)e * NHEAD + h] /
                  fmaxf(denom[(size_t)d * NHEAD + h], 1e-16f);
    f32x4 vv = *(const f32x4*)(vb + (size_t)s * HDIM + c * 4);
    float* o = agg + (size_t)d * HDIM + c * 4;
    atomicAdd(o + 0, alpha * vv.x);
    atomicAdd(o + 1, alpha * vv.y);
    atomicAdd(o + 2, alpha * vv.z);
    atomicAdd(o + 3, alpha * vv.w);
}

// ---------------------------------------------------------------------------
// Pooling + output head
// ---------------------------------------------------------------------------
__global__ void __launch_bounds__(256)
pool_accum_kernel(const float* __restrict__ h, const int* __restrict__ batch,
                  float* __restrict__ pooled, float* __restrict__ counts, int n)
{
    size_t i = (size_t)blockIdx.x * 256 + threadIdx.x;
    if (i >= (size_t)n * HDIM) return;
    int node = (int)(i >> 9);
    int dim  = (int)(i & 511);
    int g = batch[node];
    atomicAdd(&pooled[(size_t)g * HDIM + dim], h[i]);
    if (dim == 0) atomicAdd(&counts[g], 1.0f);
}

__global__ void __launch_bounds__(256)
final_linear_kernel(const float* __restrict__ pooled, const float* __restrict__ counts,
                    const float* __restrict__ Wl, const float* __restrict__ bl,
                    float* __restrict__ out)
{
    int t = blockIdx.x * 256 + threadIdx.x;
    if (t >= NGRAPH * NOUT) return;
    int g = t / NOUT, o = t % NOUT;
    float inv = 1.0f / fmaxf(counts[g], 1.0f);
    float acc = bl[o];
    const float* p = pooled + (size_t)g * HDIM;
    for (int j = 0; j < HDIM; ++j) acc += p[j] * inv * Wl[(size_t)j * NOUT + o];
    out[t] = acc;
}

// ---------------------------------------------------------------------------
// Host-side orchestration
// ---------------------------------------------------------------------------
extern "C" void kernel_launch(void* const* d_in, const int* in_sizes, int n_in,
                              void* d_out, int out_size, void* d_ws, size_t ws_size,
                              hipStream_t stream)
{
    const float* x     = (const float*)d_in[0];
    const int*   ei    = (const int*)  d_in[1];
    const int*   batch = (const int*)  d_in[2];
    const float* Wq1 = (const float*)d_in[3];  const float* bq1 = (const float*)d_in[4];
    const float* Wk1 = (const float*)d_in[5];  const float* bk1 = (const float*)d_in[6];
    const float* Wv1 = (const float*)d_in[7];  const float* bv1 = (const float*)d_in[8];
    const float* Ws1 = (const float*)d_in[9];  const float* bs1 = (const float*)d_in[10];
    const float* Wq2 = (const float*)d_in[11]; const float* bq2 = (const float*)d_in[12];
    const float* Wk2 = (const float*)d_in[13]; const float* bk2 = (const float*)d_in[14];
    const float* Wv2 = (const float*)d_in[15]; const float* bv2 = (const float*)d_in[16];
    const float* Ws2 = (const float*)d_in[17]; const float* bs2 = (const float*)d_in[18];
    const float* Wl  = (const float*)d_in[19]; const float* bl  = (const float*)d_in[20];
    float* out = (float*)d_out;

    const int N_ = in_sizes[0] / 128;       // 20000
    const int E_ = in_sizes[1] / 2;         // 320000
    const int* src = ei;
    const int* dst = ei + E_;
    const int nRowTiles = (N_ + 15) / 16;   // 1250 (N multiple of 16)

    // --- workspace carve-out -------------------------------------------------
    char* ws = (char*)d_ws;
    size_t off = 0;
    auto carve = [&](size_t bytes) -> char* {
        char* p = ws + off;
        off += (bytes + 255) & ~(size_t)255;
        return p;
    };
    unsigned short* XB = (unsigned short*)carve((size_t)N_ * HDIM * 2);   // bf16 activations
    unsigned short* WT1[4], *WT2[4];
    for (int i = 0; i < 4; ++i) WT1[i] = (unsigned short*)carve((size_t)HDIM * 128 * 2);
    for (int i = 0; i < 4; ++i) WT2[i] = (unsigned short*)carve((size_t)HDIM * HDIM * 2);
    float* QB = (float*)carve((size_t)N_ * HDIM * 4);
    float* KB = (float*)carve((size_t)N_ * HDIM * 4);
    float* VB = (float*)carve((size_t)N_ * HDIM * 4);
    float* SB = (float*)carve((size_t)N_ * HDIM * 4);   // skip, then skip+agg
    float* LG = (float*)carve((size_t)E_ * NHEAD * 4);  // logits -> ex (in place)
    float* MB = (float*)carve((size_t)N_ * NHEAD * 4);
    float* DN = (float*)carve((size_t)N_ * NHEAD * 4);
    float* PL = (float*)carve((size_t)NGRAPH * HDIM * 4);
    float* CT = (float*)carve((size_t)NGRAPH * 4);
    (void)ws_size; (void)n_in; (void)out_size;

    // --- precompute: bf16 weights (transposed) + bf16 input ------------------
    {
        int n1 = 128 * HDIM, g1 = (n1 + 255) / 256;
        transpose_bf16_kernel<<<g1, 256, 0, stream>>>(Wq1, WT1[0], 128, HDIM);
        transpose_bf16_kernel<<<g1, 256, 0, stream>>>(Wk1, WT1[1], 128, HDIM);
        transpose_bf16_kernel<<<g1, 256, 0, stream>>>(Wv1, WT1[2], 128, HDIM);
        transpose_bf16_kernel<<<g1, 256, 0, stream>>>(Ws1, WT1[3], 128, HDIM);
        int n2 = HDIM * HDIM, g2 = (n2 + 255) / 256;
        transpose_bf16_kernel<<<g2, 256, 0, stream>>>(Wq2, WT2[0], HDIM, HDIM);
        transpose_bf16_kernel<<<g2, 256, 0, stream>>>(Wk2, WT2[1], HDIM, HDIM);
        transpose_bf16_kernel<<<g2, 256, 0, stream>>>(Wv2, WT2[2], HDIM, HDIM);
        transpose_bf16_kernel<<<g2, 256, 0, stream>>>(Ws2, WT2[3], HDIM, HDIM);
        size_t nx = (size_t)N_ * 128;
        cvt_bf16_kernel<<<(unsigned)((nx + 255) / 256), 256, 0, stream>>>(x, XB, nullptr, nx, 0);
    }

    const int EH = E_ * NHEAD;
    const dim3 gemmGrid((nRowTiles + 7) / 8, HDIM / 64);

    auto run_layer = [&](int K, unsigned short* const* WT,
                         const float* bq, const float* bk, const float* bv, const float* bs,
                         unsigned short* outBf, float* outF) {
        // 4 projections (WMMA bf16)
        gemm_bf16_wmma<<<gemmGrid, 256, 0, stream>>>(XB, WT[0], bq, QB, K, nRowTiles);
        gemm_bf16_wmma<<<gemmGrid, 256, 0, stream>>>(XB, WT[1], bk, KB, K, nRowTiles);
        gemm_bf16_wmma<<<gemmGrid, 256, 0, stream>>>(XB, WT[2], bv, VB, K, nRowTiles);
        gemm_bf16_wmma<<<gemmGrid, 256, 0, stream>>>(XB, WT[3], bs, SB, K, nRowTiles);
        // segment-softmax scratch init
        size_t nh = (size_t)N_ * NHEAD;
        fill_f32_kernel<<<(unsigned)((nh + 255) / 256), 256, 0, stream>>>(MB, -3.402823466e38f, nh);
        fill_f32_kernel<<<(unsigned)((nh + 255) / 256), 256, 0, stream>>>(DN, 0.0f, nh);
        // attention
        edge_logits_kernel<<<(E_ + 7) / 8, 256, 0, stream>>>(QB, KB, src, dst, LG, E_);
        seg_max_kernel<<<(EH + 255) / 256, 256, 0, stream>>>(LG, dst, MB, EH);
        exp_denom_kernel<<<(EH + 255) / 256, 256, 0, stream>>>(LG, dst, MB, DN, EH);
        long long tot = (long long)E_ * 128;
        scatter_kernel<<<(unsigned)((tot + 255) / 256), 256, 0, stream>>>(LG, DN, VB, src, dst, SB, tot);
        // relu(agg + skip) -> bf16 (next layer) and/or f32 (pooling)
        size_t nn = (size_t)N_ * HDIM;
        cvt_bf16_kernel<<<(unsigned)((nn + 255) / 256), 256, 0, stream>>>(SB, outBf, outF, nn, 1);
    };

    unsigned short* wt1p[4] = { WT1[0], WT1[1], WT1[2], WT1[3] };
    unsigned short* wt2p[4] = { WT2[0], WT2[1], WT2[2], WT2[3] };
    run_layer(128,  wt1p, bq1, bk1, bv1, bs1, XB, nullptr);   // h1 -> XB (bf16)
    run_layer(HDIM, wt2p, bq2, bk2, bv2, bs2, nullptr, QB);   // h2 -> QB (f32)

    // --- global mean pool + output linear ------------------------------------
    size_t np = (size_t)NGRAPH * HDIM;
    fill_f32_kernel<<<(unsigned)((np + 255) / 256), 256, 0, stream>>>(PL, 0.0f, np);
    fill_f32_kernel<<<1, 256, 0, stream>>>(CT, 0.0f, (size_t)NGRAPH);
    size_t nh2 = (size_t)N_ * HDIM;
    pool_accum_kernel<<<(unsigned)((nh2 + 255) / 256), 256, 0, stream>>>(QB, batch, PL, CT, N_);
    final_linear_kernel<<<(NGRAPH * NOUT + 255) / 256, 256, 0, stream>>>(PL, CT, Wl, bl, out);
}
